// MyLayer_3298534884153
// MI455X (gfx1250) — matrix operand order repaired and checked by reading, not compiled, not run on hardware
//
#include <hip/hip_runtime.h>
#include <stdint.h>

typedef __attribute__((ext_vector_type(16))) _Float16 v16h;
typedef __attribute__((ext_vector_type(8)))  _Float16 v8h;
typedef __attribute__((ext_vector_type(4)))  _Float16 v4h;
typedef __attribute__((ext_vector_type(8)))  float    v8f;
typedef __attribute__((ext_vector_type(4)))  float    v4f;
typedef __attribute__((ext_vector_type(4)))  unsigned int v4u;
typedef __attribute__((ext_vector_type(8)))  int      v8i;
typedef __attribute__((ext_vector_type(4)))  int      v4i;

#define M_DIM 64
#define K_DIM 4096
#define N_DIM 14336
#define GRP   128              // scale group size along K
#define KW    (K_DIM / 8)      // 512 packed int32 per weight row
#define NGRP  (K_DIM / GRP)    // 32 scale groups
#define NSTEP (K_DIM / 32)     // 128 k-steps of 32

// ---------------- pass 1: convert x fp32 -> fp16 into workspace ----------------
__global__ __launch_bounds__(256) void xcvt_kernel(const float* __restrict__ x,
                                                   _Float16* __restrict__ xh) {
  int i = blockIdx.x * blockDim.x + threadIdx.x;   // one v4 per thread
  v4f v = ((const v4f* __restrict__)x)[i];
  ((v4h* __restrict__)xh)[i] = __builtin_convertvector(v, v4h);
}

// ---------------- in-register int4 -> f16 dequant (pack-perm aware) ------------
// Column pair (2p,2p+1) = nibbles (p,p+4):  ((w>>4p)&0x000F000F)|0x64006400 gives
// packed f16 (1024+q_lo,1024+q_hi) in column order; -1032 (exact) yields q-8.
__device__ __forceinline__ v16h dq16(uint32_t w0, uint32_t w1) {
  union { uint32_t u[8]; v16h h; } r;
#pragma unroll
  for (int p = 0; p < 4; ++p) {
    r.u[p]     = ((w0 >> (4 * p)) & 0x000F000Fu) | 0x64006400u;
    r.u[p + 4] = ((w1 >> (4 * p)) & 0x000F000Fu) | 0x64006400u;
  }
  v16h bias;
#pragma unroll
  for (int i = 0; i < 16; ++i) bias[i] = (_Float16)1032.0f;
  return r.h - bias;
}

// A-operand fetch from the LDS slab: two ds_load_b128 per 16x32 tile row
__device__ __forceinline__ v16h ld_a(const _Float16* lp, int row) {
  union { v16h h; v8h p[2]; } a;
  a.p[0] = *(const v8h*)(lp + row * GRP);
  a.p[1] = *(const v8h*)(lp + row * GRP + 8);
  return a.h;
}

// ---------------- TDM: stage one 64x128 f16 A-slab into LDS --------------------
// D# per CDNA5 ISA:
//   group0: count=1 | lds_addr | global_addr[56:0] | type=2
//   group1: data_size=2B, tensor_dim0=4096 (stride 4096), tensor_dim1=64,
//           tile_dim0=128, tile_dim1=64   (2-D tile -> groups 2/3/4 zero)
__device__ __forceinline__ void tdm_load_group(const _Float16* xh,
                                               unsigned lds_off, int g) {
  unsigned long long ga = (unsigned long long)(uintptr_t)(xh + (size_t)g * GRP);
  v4u g0;
  g0.x = 0x1u;                                    // count=1, user descriptor
  g0.y = lds_off;                                 // lds_addr (bytes)
  g0.z = (unsigned)(ga & 0xFFFFFFFFu);            // global_addr[31:0]
  g0.w = (unsigned)((ga >> 32) & 0x01FFFFFFu)     // global_addr[56:32]
         | (2u << 30);                            // type = 2 ("image")
  v8i g1;
  g1[0] = 0x00010000;                             // data_size=1 -> 2 bytes
  g1[1] = (int)(K_DIM << 16);                     // tensor_dim0[15:0]
  g1[2] = (int)((unsigned)(K_DIM >> 16) | ((unsigned)M_DIM << 16)); // dim0 hi|dim1 lo
  g1[3] = (int)(GRP << 16);                       // tile_dim0=128
  g1[4] = M_DIM;                                  // tile_dim1=64
  g1[5] = K_DIM;                                  // tensor_dim0_stride lo32
  g1[6] = 0;                                      // stride hi | dim1_stride lo
  g1[7] = 0;
  v4i z4 = {0, 0, 0, 0};
  v8i z8 = {0, 0, 0, 0, 0, 0, 0, 0};
  __builtin_amdgcn_tensor_load_to_lds(g0, g1, z4, z4, z8, 0);
}

// ---------------- pass 2: W4A16 GEMM -------------------------------------------
// Block = 256 threads (8 waves).  Wave w owns n-tile blockIdx*8+w and computes
// ALL 4 m-tiles (64x16 output): one dequantized B feeds 4 WMMAs.
// A slab staged in LDS by TDM (double-buffered); A operands ping-pong (even/odd
// step register sets); B words flow through a 3-deep queue with the dequant
// itself pipelined one step ahead (kills the WMMA->VALU WAR on B registers).
__global__ __launch_bounds__(256) void w4a16_gemm_kernel(
    const _Float16* __restrict__ xh,       // (64, 4096) f16
    const uint32_t* __restrict__ Bp,       // (14336, 512) packed int4
    const float*    __restrict__ s,        // (32, 14336) scales
    float*          __restrict__ out) {    // (64, 14336)
  __shared__ _Float16 smem[2][M_DIM * GRP];              // 2 x 16 KB

  const int lane = threadIdx.x & 31;
  const int w    = threadIdx.x >> 5;                     // wave in block
  const int l16  = lane & 15;
  const int hi   = lane >> 4;
  const int n    = blockIdx.x * 128 + w * 16 + l16;      // output column

  const uint32_t* bptr = Bp + n * KW + 2 * hi;
  const unsigned lds0 = (unsigned)(uintptr_t)&smem[0][0];
  const unsigned lds1 = (unsigned)(uintptr_t)&smem[1][0];

  // prologue: issue TDM for group 0 (one wave; EXEC ignored by TDM)
  if (threadIdx.x < 32) tdm_load_group(xh, lds0, 0);

  // B pipeline: bC = dequantized B for current step; wq0/wq1 = raw words for
  // steps +1/+2; loads issued 3 steps ahead.
  uint2 wd0 = *(const uint2*)(bptr);                     // ks = 0
  uint2 wq0 = *(const uint2*)(bptr + 4);                 // ks = 1
  uint2 wq1 = *(const uint2*)(bptr + 8);                 // ks = 2
  v16h bC = dq16(wd0.x, wd0.y);

  v8f acc0 = {}, acc1 = {}, acc2 = {}, acc3 = {};

  for (int g = 0; g < NGRP; ++g) {
    const int buf = g & 1;
    if (threadIdx.x < 32) {
      if (g + 1 < NGRP) {
        tdm_load_group(xh, (g + 1) & 1 ? lds1 : lds0, g + 1);
        __builtin_amdgcn_s_wait_tensorcnt(1);   // group g landed; g+1 in flight
      } else {
        __builtin_amdgcn_s_wait_tensorcnt(0);
      }
    }
    __syncthreads();                            // A slab for g visible to all

    const float sg = s[g * N_DIM + n];
    __builtin_prefetch(bptr + (g + 2) * (GRP / 8), 0, 0);

    const _Float16* sb = &smem[buf][0] + 16 * hi;        // + t*32 + row*GRP

    // A operands, even/odd ping-pong register sets (no rotation copies)
    v16h a0[4], a1[4];
#pragma unroll
    for (int mt = 0; mt < 4; ++mt) a0[mt] = ld_a(sb, mt * 16 + l16);

    v8f c0 = {}, c1 = {}, c2 = {}, c3 = {};     // unscaled group accumulators
#pragma unroll
    for (int t = 0; t < 4; ++t) {               // 4 k-steps of 32
      v16h* aC = (t & 1) ? a1 : a0;
      v16h* aN = (t & 1) ? a0 : a1;
      if (t < 3) {                              // prefetch next step's A
#pragma unroll
        for (int mt = 0; mt < 4; ++mt)
          aN[mt] = ld_a(sb + (t + 1) * 32, mt * 16 + l16);
      }
      // B: issue load 3 steps ahead, dequant 1 step ahead (distinct regs)
      const int ks = g * 4 + t;
      const int nx = (ks + 3 < NSTEP) ? ks + 3 : NSTEP - 1;
      uint2 wdN = *(const uint2*)(bptr + nx * 4);
      v16h bN = dq16(wq0.x, wq0.y);             // B for step ks+1

      c0 = __builtin_amdgcn_wmma_f32_16x16x32_f16(false, aC[0], false, bC,
                                                  (short)0, c0, false, false);
      c1 = __builtin_amdgcn_wmma_f32_16x16x32_f16(false, aC[1], false, bC,
                                                  (short)0, c1, false, false);
      c2 = __builtin_amdgcn_wmma_f32_16x16x32_f16(false, aC[2], false, bC,
                                                  (short)0, c2, false, false);
      c3 = __builtin_amdgcn_wmma_f32_16x16x32_f16(false, aC[3], false, bC,
                                                  (short)0, c3, false, false);

      bC = bN; wq0 = wq1; wq1 = wdN;            // rotate B pipeline
    }

    // fold per-(group, n) scale once per 128 K in fp32
#pragma unroll
    for (int r = 0; r < 8; ++r) {
      acc0[r] += sg * c0[r];
      acc1[r] += sg * c1[r];
      acc2[r] += sg * c2[r];
      acc3[r] += sg * c3[r];
    }
    __syncthreads();                 // all waves done with buf before TDM refill
  }

  // C/D layout: VGPR r holds row (r + 8*hi), column = lane&15
#pragma unroll
  for (int r = 0; r < 8; ++r) {
    out[(0 * 16 + 8 * hi + r) * N_DIM + n] = acc0[r];
    out[(1 * 16 + 8 * hi + r) * N_DIM + n] = acc1[r];
    out[(2 * 16 + 8 * hi + r) * N_DIM + n] = acc2[r];
    out[(3 * 16 + 8 * hi + r) * N_DIM + n] = acc3[r];
  }
}

// -------------------------------------------------------------------------------
extern "C" void kernel_launch(void* const* d_in, const int* in_sizes, int n_in,
                              void* d_out, int out_size, void* d_ws, size_t ws_size,
                              hipStream_t stream) {
  const float*    x   = (const float*)d_in[0];     // (64, 4096) fp32
  const uint32_t* Bp  = (const uint32_t*)d_in[1];  // (14336, 512) int32
  const float*    s   = (const float*)d_in[2];     // (32, 14336) fp32
  float*          out = (float*)d_out;             // (64, 14336) fp32
  _Float16*       xh  = (_Float16*)d_ws;           // 512 KB f16 copy of x

  // pass 1: x -> f16  (262144 elems, 4 per thread)
  xcvt_kernel<<<(M_DIM * K_DIM) / (256 * 4), 256, 0, stream>>>(x, xh);

  // pass 2: 112 blocks x 256 threads; wave = 64x16 output tile over full K
  const int blocks = N_DIM / 128;                  // 112
  w4a16_gemm_kernel<<<blocks, 256, 0, stream>>>(xh, Bp, s, out);
}
